// MultiHeadAtt_19121194401991
// MI455X (gfx1250) — compile-verified, matrix operand order checked by reading, not run on hardware
//
#include <hip/hip_runtime.h>

typedef __bf16 bf16_t;
typedef __attribute__((ext_vector_type(16))) __bf16 v16bf;
typedef __attribute__((ext_vector_type(8)))  float  v8f;

#ifndef __has_builtin
#define __has_builtin(x) 0
#endif

// Problem constants (Transformer-XL rel-attn)
constexpr int QLEN = 512, MLEN = 512, KLEN = 1024, RLEN = 1024;
constexpr int BSZ = 8, NH = 16, DH = 64, DM = 1024, QKV3 = 3072, NHD = 1024;

__device__ __forceinline__ bf16_t f2bf(float f) {
  unsigned u = __float_as_uint(f);
  u += 0x7fffu + ((u >> 16) & 1u);  // round-to-nearest-even
  unsigned short h = (unsigned short)(u >> 16);
  bf16_t o;
  __builtin_memcpy(&o, &h, 2);
  return o;
}
__device__ __forceinline__ float bf2f(bf16_t x) {
  unsigned short h;
  __builtin_memcpy(&h, &x, 2);
  return __uint_as_float((unsigned)h << 16);
}

// ---------------------------------------------------------------------------
// CDNA5 async copy global->LDS (ASYNCcnt-tracked, overlaps with WMMA).
// Signature probe-derived: (AS1 vNi* src, AS3 vNi* dst, imm offset, imm cpol).
// ---------------------------------------------------------------------------
#if __has_builtin(__builtin_amdgcn_global_load_async_to_lds_b128)
#define ASYNC_CHUNK 8  // bf16 elements per async op
typedef __attribute__((__vector_size__(16))) int v4i_t;
__device__ __forceinline__ void async_chunk(const bf16_t* g, bf16_t* l) {
  typedef __attribute__((address_space(1))) v4i_t gv_t;
  typedef __attribute__((address_space(3))) v4i_t lv_t;
  __builtin_amdgcn_global_load_async_to_lds_b128(
      (gv_t*)(unsigned long long)(const void*)g,
      (lv_t*)(unsigned)(unsigned long long)(void*)l, 0, 0);
}
#elif __has_builtin(__builtin_amdgcn_global_load_async_to_lds_b64)
#define ASYNC_CHUNK 4
typedef __attribute__((__vector_size__(8))) int v2i_t;
__device__ __forceinline__ void async_chunk(const bf16_t* g, bf16_t* l) {
  typedef __attribute__((address_space(1))) v2i_t gv_t;
  typedef __attribute__((address_space(3))) v2i_t lv_t;
  __builtin_amdgcn_global_load_async_to_lds_b64(
      (gv_t*)(unsigned long long)(const void*)g,
      (lv_t*)(unsigned)(unsigned long long)(void*)l, 0, 0);
}
#else
#define ASYNC_CHUNK 4
__device__ __forceinline__ void async_chunk(const bf16_t* g, bf16_t* l) {
  unsigned lds = (unsigned)(unsigned long long)(void*)l;
  unsigned long long ga = (unsigned long long)(const void*)g;
  asm volatile("global_load_async_to_lds_b64 %0, %1, off" ::"v"(lds), "v"(ga)
               : "memory");
}
#endif

__device__ __forceinline__ void wait_async_0() {
#if __has_builtin(__builtin_amdgcn_s_wait_asynccnt)
  __builtin_amdgcn_s_wait_asynccnt(0);
#else
  asm volatile("s_wait_asynccnt 0x0" ::: "memory");
#endif
}

// ---------------------------------------------------------------------------
// Elementwise conversion / packing kernels
// ---------------------------------------------------------------------------
__global__ __launch_bounds__(256)
void f32_to_bf16_kernel(const float* __restrict__ in, bf16_t* __restrict__ out, long n) {
  long i = (long)blockIdx.x * 256 + threadIdx.x;
  if (i < n) out[i] = f2bf(in[i]);
}

__global__ __launch_bounds__(256)
void cat_to_bf16_kernel(const float* __restrict__ mems, const float* __restrict__ w,
                        bf16_t* __restrict__ out) {
  long i = (long)blockIdx.x * 256 + threadIdx.x;
  constexpr long HALF = (long)MLEN * BSZ * DM;
  constexpr long TOT  = (long)KLEN * BSZ * DM;
  if (i < TOT) out[i] = f2bf(i < HALF ? mems[i] : w[i - HALF]);
}

// in [R][C] f32 -> out [C][R] bf16 (tiled via LDS; R,C multiples of 32)
__global__ __launch_bounds__(256)
void transpose_f32_to_bf16(const float* __restrict__ in, bf16_t* __restrict__ out,
                           int R, int C) {
  __shared__ float tile[32][33];
  const int bx = blockIdx.x * 32, by = blockIdx.y * 32;
  const int tx = threadIdx.x & 31, ty = threadIdx.x >> 5;  // 32x8
#pragma unroll
  for (int yy = 0; yy < 32; yy += 8)
    tile[ty + yy][tx] = in[(long)(by + ty + yy) * C + (bx + tx)];
  __syncthreads();
#pragma unroll
  for (int yy = 0; yy < 32; yy += 8)
    out[(long)(bx + ty + yy) * R + (by + tx)] = f2bf(tile[tx][ty + yy]);
}

// w_heads bf16 [8192][3072] (row = j*8+b) ->
//   qw/qr [b][n][i][d] (+bias),  k [b][n][j][d],  v TRANSPOSED [b][n][d][j]
__global__ __launch_bounds__(256)
void pack_qkv_kernel(const bf16_t* __restrict__ wh,
                     const float* __restrict__ rwb, const float* __restrict__ rrb,
                     bf16_t* __restrict__ qw, bf16_t* __restrict__ qr,
                     bf16_t* __restrict__ kk, bf16_t* __restrict__ vv) {
  long idx = (long)blockIdx.x * 256 + threadIdx.x;
  constexpr long TOT = (long)KLEN * BSZ * NHD;
  if (idx >= TOT) return;
  int h = (int)(idx % NHD);
  long t = idx / NHD;
  int b = (int)(t % BSZ);
  int j = (int)(t / BSZ);
  int n = h >> 6, d = h & 63;
  const bf16_t* src = wh + ((long)j * BSZ + b) * QKV3;
  kk[(((long)(b * NH + n) * KLEN) + j) * DH + d] = src[1024 + h];
  vv[(((long)(b * NH + n) * DH) + d) * KLEN + j] = src[2048 + h];
  if (j >= MLEN) {
    int i = j - MLEN;
    long qdst = (((long)(b * NH + n) * QLEN) + i) * DH + d;
    float q = bf2f(src[h]);
    qw[qdst] = f2bf(q + rwb[h]);
    qr[qdst] = f2bf(q + rrb[h]);
  }
}

// r_head_k bf16 [1024][1024] -> rk_bf [n][j][d]
__global__ __launch_bounds__(256)
void pack_rk_kernel(const bf16_t* __restrict__ rh, bf16_t* __restrict__ rkb) {
  long i = (long)blockIdx.x * 256 + threadIdx.x;
  if (i >= (long)RLEN * NHD) return;
  int h = (int)(i % NHD);
  int j = (int)(i / NHD);
  int n = h >> 6, d = h & 63;
  rkb[((long)n * RLEN + j) * DH + d] = rh[i];
}

// ---------------------------------------------------------------------------
// bf16 WMMA GEMM:  C(MxN) = A(MxK,bf16) * B^T  with B stored [N][K].
// Double-buffered LDS, async global->LDS staging overlapped with WMMA.
//   EPI: 0 = f32 store, 1 = f32 store + resid add, 3 = bf16 store
// ---------------------------------------------------------------------------
template <int BM, int BN, int WR, int WC, int EPI>
__global__ __launch_bounds__(WR * WC * 32)
void gemm_bf16_wmma(const bf16_t* __restrict__ A, const bf16_t* __restrict__ B,
                    float* __restrict__ C, const float* __restrict__ resid,
                    int K, int lda, int ldb, int ldc,
                    long sA, long sB, int bMod, int zDiv, long cOuter, long cInner) {
  constexpr int BK = 32;
  constexpr int NT = WR * WC * 32;
  constexpr int IM = BM / (WR * 16);
  constexpr int IN = BN / (WC * 16);
  constexpr int LDSS = BK + 8;           // 80B row stride (16B multiple)
  constexpr int CPR = BK / ASYNC_CHUNK;  // async chunks per tile row
  constexpr int APC = (BM * CPR) / NT;
  constexpr int BPC = (BN * CPR) / NT;

  __shared__ __align__(16) bf16_t As[2][BM][LDSS];
  __shared__ __align__(16) bf16_t Bs[2][BN][LDSS];

  const int z = blockIdx.z;
  A += (long)z * sA;
  B += (long)(bMod ? (z % bMod) : z) * sB;
  const long cOff = (long)(z / zDiv) * cOuter + (long)(z % zDiv) * cInner;
  float*  Cf = C + cOff;             // EPI 0/1
  bf16_t* Cb = (bf16_t*)C + cOff;    // EPI 3

  const int tid  = threadIdx.x;
  const int wv   = tid >> 5;
  const int lane = tid & 31;
  const int wr   = wv / WC;
  const int wc   = wv % WC;
  const int hf   = lane >> 4;
  const int l16  = lane & 15;
  const int bm0  = blockIdx.y * BM;
  const int bn0  = blockIdx.x * BN;

  v8f acc[IM][IN];
#pragma unroll
  for (int im = 0; im < IM; ++im)
#pragma unroll
    for (int in = 0; in < IN; ++in)
#pragma unroll
      for (int e = 0; e < 8; ++e) acc[im][in][e] = 0.0f;

  auto stage = [&](int buf, int k0) {
#pragma unroll
    for (int p = 0; p < APC; ++p) {
      int idx = p * NT + tid;
      int row = idx / CPR;
      int cc  = (idx % CPR) * ASYNC_CHUNK;
      async_chunk(A + (long)(bm0 + row) * lda + (k0 + cc), &As[buf][row][cc]);
    }
#pragma unroll
    for (int p = 0; p < BPC; ++p) {
      int idx = p * NT + tid;
      int row = idx / CPR;
      int cc  = (idx % CPR) * ASYNC_CHUNK;
      async_chunk(B + (long)(bn0 + row) * ldb + (k0 + cc), &Bs[buf][row][cc]);
    }
  };

  auto compute = [&](int buf) {
    v16bf af[IM], bg[IN];
#pragma unroll
    for (int im = 0; im < IM; ++im) {
      const bf16_t* ap = &As[buf][wr * IM * 16 + im * 16 + l16][hf * 8];
#pragma unroll
      for (int e = 0; e < 8; ++e) { af[im][e] = ap[e]; af[im][e + 8] = ap[16 + e]; }
    }
#pragma unroll
    for (int in = 0; in < IN; ++in) {
      const bf16_t* bp = &Bs[buf][wc * IN * 16 + in * 16 + l16][hf * 8];
#pragma unroll
      for (int e = 0; e < 8; ++e) { bg[in][e] = bp[e]; bg[in][e + 8] = bp[16 + e]; }
    }
#pragma unroll
    for (int im = 0; im < IM; ++im)
#pragma unroll
      for (int in = 0; in < IN; ++in)
        acc[im][in] = __builtin_amdgcn_wmma_f32_16x16x32_bf16(
            false, af[im], false, bg[in], (short)0, acc[im][in], false, false);
  };

  const int nk = K / BK;
  stage(0, 0);
  wait_async_0();
  __syncthreads();
  for (int kt = 0; kt < nk; ++kt) {
    const int cur = kt & 1;
    if (kt + 1 < nk) stage(cur ^ 1, (kt + 1) * BK);  // DMA overlaps WMMA
    compute(cur);
    if (kt + 1 < nk) {
      wait_async_0();
      __syncthreads();
    }
  }

#pragma unroll
  for (int im = 0; im < IM; ++im) {
#pragma unroll
    for (int in = 0; in < IN; ++in) {
      const int col = bn0 + wc * IN * 16 + in * 16 + l16;
      const int mb  = bm0 + wr * IM * 16 + im * 16 + hf * 8;
#pragma unroll
      for (int e = 0; e < 8; ++e) {
        const int row = mb + e;
        const float val = acc[im][in][e];
        const long ci = (long)row * ldc + col;
        if (EPI == 0)      Cf[ci] = val;
        else if (EPI == 1) Cf[ci] = val + resid[ci];
        else               Cb[ci] = f2bf(val);
      }
    }
  }
}

// ---------------------------------------------------------------------------
// Fused rel-attention (flash-style, online softmax):
//   S = qw@K^T ; S += gather(BD_raw, rel_shift) ; *=scale ; mask
//   O = softmax(S) @ V   (P staged bf16 in LDS, V async-staged, all WMMA)
// rel_shift identity: shift(BD)[i][j] = BDr[i][511+j-i]      (j-i <= 512)
//                                     = 0                     (j-i == 513)
//                                     = BDr[i+1][j-i-514]     (j-i >  513)
// Gather is branchless: the c==1024 case reads (i+1)*1024-1 (always in
// bounds because c==1024 implies i<=510) and is select-zeroed afterwards.
// ---------------------------------------------------------------------------
__global__ __launch_bounds__(256)
void attn_fused_kernel(const bf16_t* __restrict__ qw,   // [z][512][64]
                       const bf16_t* __restrict__ kk,   // [z][1024][64]
                       const bf16_t* __restrict__ vt,   // [z][64][1024]
                       const bf16_t* __restrict__ bdr,  // [z][512][1024] bf16
                       const int* __restrict__ mask,    // [512][8]
                       bf16_t* __restrict__ av) {       // [512][8][1024]
  constexpr int BQ = 128, BJ = 128;
  constexpr int IM = 2, INS = 4, INO = 2;
  constexpr int CH = ASYNC_CHUNK;
  constexpr int QC = (BQ * DH / CH) / 256;  // per-thread chunks, q/K tile
  constexpr int VC = (DH * BJ / CH) / 256;  // per-thread chunks, V tile
  constexpr float scale = 0.125f;

  __shared__ __align__(16) bf16_t Aq[BQ][72];        // q block (K=64 + pad 8)
  __shared__ __align__(16) bf16_t Kt[2][BJ][72];
  __shared__ __align__(16) bf16_t Vt[2][DH][136];    // [d][j] (+pad 8)
  __shared__ __align__(16) bf16_t Pt[BQ][136];
  __shared__ float rmax[2][BQ];
  __shared__ float rsum[2][BQ];

  const int z = blockIdx.y, b = z >> 4, n = z & 15;
  const int i0 = blockIdx.x * BQ;
  qw  += (long)z * QLEN * DH;
  kk  += (long)z * KLEN * DH;
  vt  += (long)z * DH * KLEN;
  bdr += (long)z * QLEN * RLEN;

  const int tid = threadIdx.x, wv = tid >> 5, lane = tid & 31;
  const int wr = wv >> 1, wc = wv & 1, hf = lane >> 4, l16 = lane & 15;

  auto stageK = [&](int buf, int j0) {
#pragma unroll
    for (int p = 0; p < QC; ++p) {
      int idx = p * 256 + tid, row = idx / (DH / CH), cc = (idx % (DH / CH)) * CH;
      async_chunk(kk + (long)(j0 + row) * DH + cc, &Kt[buf][row][cc]);
    }
  };
  auto stageV = [&](int buf, int j0) {
#pragma unroll
    for (int p = 0; p < VC; ++p) {
      int idx = p * 256 + tid, row = idx / (BJ / CH), cc = (idx % (BJ / CH)) * CH;
      async_chunk(vt + (long)row * KLEN + (j0 + cc), &Vt[buf][row][cc]);
    }
  };
#pragma unroll
  for (int p = 0; p < QC; ++p) {  // q block: staged once
    int idx = p * 256 + tid, row = idx / (DH / CH), cc = (idx % (DH / CH)) * CH;
    async_chunk(qw + (long)(i0 + row) * DH + cc, &Aq[row][cc]);
  }
  stageK(0, 0);
  stageV(0, 0);
  wait_async_0();
  __syncthreads();

  float mrow[IM][8], lrow[IM][8];
  bool  mskd[IM][8];
  v8f acco[IM][INO];
#pragma unroll
  for (int im = 0; im < IM; ++im)
#pragma unroll
    for (int e = 0; e < 8; ++e) {
      mrow[im][e] = -3.0e38f;
      lrow[im][e] = 0.0f;
      mskd[im][e] = mask[(long)(i0 + wr * 32 + im * 16 + hf * 8 + e) * BSZ + b] != 0;
    }
#pragma unroll
  for (int im = 0; im < IM; ++im)
#pragma unroll
    for (int io = 0; io < INO; ++io)
#pragma unroll
      for (int e = 0; e < 8; ++e) acco[im][io][e] = 0.0f;

  for (int jt = 0; jt < KLEN / BJ; ++jt) {
    const int j0 = jt * BJ;
    const int cur = jt & 1;
    if (jt + 1 < KLEN / BJ) {  // prefetch next K/V tiles (overlaps everything)
      stageK(cur ^ 1, j0 + BJ);
      stageV(cur ^ 1, j0 + BJ);
    }
    // ---- S = qw @ K^T (K=64: two 32-k-steps)
    v8f s[IM][INS];
#pragma unroll
    for (int im = 0; im < IM; ++im)
#pragma unroll
      for (int in = 0; in < INS; ++in)
#pragma unroll
        for (int e = 0; e < 8; ++e) s[im][in][e] = 0.0f;
#pragma unroll
    for (int ks = 0; ks < 2; ++ks) {
      v16bf af[IM], bg[INS];
#pragma unroll
      for (int im = 0; im < IM; ++im) {
        const bf16_t* ap = &Aq[wr * 32 + im * 16 + l16][ks * 32 + hf * 8];
#pragma unroll
        for (int e = 0; e < 8; ++e) { af[im][e] = ap[e]; af[im][e + 8] = ap[16 + e]; }
      }
#pragma unroll
      for (int in = 0; in < INS; ++in) {
        const bf16_t* bp = &Kt[cur][wc * 64 + in * 16 + l16][ks * 32 + hf * 8];
#pragma unroll
        for (int e = 0; e < 8; ++e) { bg[in][e] = bp[e]; bg[in][e + 8] = bp[16 + e]; }
      }
#pragma unroll
      for (int im = 0; im < IM; ++im)
#pragma unroll
        for (int in = 0; in < INS; ++in)
          s[im][in] = __builtin_amdgcn_wmma_f32_16x16x32_bf16(
              false, af[im], false, bg[in], (short)0, s[im][in], false, false);
    }
    // ---- BD gather (branchless, batched) + scale + mask; per-row partial max
    float pmax[IM][8];
#pragma unroll
    for (int im = 0; im < IM; ++im)
#pragma unroll
      for (int e = 0; e < 8; ++e) pmax[im][e] = -3.0e38f;
#pragma unroll
    for (int im = 0; im < IM; ++im) {
      const int ib = i0 + wr * 32 + im * 16 + hf * 8;
#pragma unroll
      for (int in = 0; in < INS; ++in) {
        const int j = j0 + wc * 64 + in * 16 + l16;
        float bdv[8];
#pragma unroll
        for (int e = 0; e < 8; ++e) {  // 8 loads clustered, one wait
          const int i = ib + e;
          const int c = 511 + j - i;
          const long off = (c < 1024) ? ((long)i * RLEN + c)
                                      : ((long)(i + 1) * RLEN + (c - 1025));
          bdv[e] = bf2f(bdr[off]);
        }
#pragma unroll
        for (int e = 0; e < 8; ++e) {
          const int c = 511 + j - (ib + e);
          const float bd = (c == 1024) ? 0.0f : bdv[e];
          float sv = (s[im][in][e] + bd) * scale;
          if (mskd[im][e]) sv = 0.0f;
          s[im][in][e] = sv;
          pmax[im][e] = fmaxf(pmax[im][e], sv);
        }
      }
    }
    // ---- row max: half-wave shuffle reduce, then cross-wave via LDS
#pragma unroll
    for (int im = 0; im < IM; ++im)
#pragma unroll
      for (int e = 0; e < 8; ++e)
#pragma unroll
        for (int o = 1; o < 16; o <<= 1)
          pmax[im][e] = fmaxf(pmax[im][e], __shfl_xor(pmax[im][e], o));
    if (l16 == 0) {
#pragma unroll
      for (int im = 0; im < IM; ++im)
#pragma unroll
        for (int e = 0; e < 8; ++e)
          rmax[wc][wr * 32 + im * 16 + hf * 8 + e] = pmax[im][e];
    }
    __syncthreads();
    float alpha[IM][8];
#pragma unroll
    for (int im = 0; im < IM; ++im)
#pragma unroll
      for (int e = 0; e < 8; ++e) {
        const float omax = rmax[wc ^ 1][wr * 32 + im * 16 + hf * 8 + e];
        const float mn = fmaxf(mrow[im][e], fmaxf(pmax[im][e], omax));
        alpha[im][e] = __expf(mrow[im][e] - mn);
        mrow[im][e] = mn;  // mrow now holds the updated max
      }
    // ---- P = exp(S - mnew): bf16 into LDS; partial row sums
    float psum[IM][8];
#pragma unroll
    for (int im = 0; im < IM; ++im)
#pragma unroll
      for (int e = 0; e < 8; ++e) psum[im][e] = 0.0f;
#pragma unroll
    for (int im = 0; im < IM; ++im)
#pragma unroll
      for (int in = 0; in < INS; ++in)
#pragma unroll
        for (int e = 0; e < 8; ++e) {
          float pv = __expf(s[im][in][e] - mrow[im][e]);
          psum[im][e] += pv;
          Pt[wr * 32 + im * 16 + hf * 8 + e][wc * 64 + in * 16 + l16] = f2bf(pv);
        }
#pragma unroll
    for (int im = 0; im < IM; ++im)
#pragma unroll
      for (int e = 0; e < 8; ++e)
#pragma unroll
        for (int o = 1; o < 16; o <<= 1) psum[im][e] += __shfl_xor(psum[im][e], o);
    if (l16 == 0) {
#pragma unroll
      for (int im = 0; im < IM; ++im)
#pragma unroll
        for (int e = 0; e < 8; ++e)
          rsum[wc][wr * 32 + im * 16 + hf * 8 + e] = psum[im][e];
    }
    // rescale O while the sum lands
#pragma unroll
    for (int im = 0; im < IM; ++im)
#pragma unroll
      for (int io = 0; io < INO; ++io)
#pragma unroll
        for (int e = 0; e < 8; ++e) acco[im][io][e] *= alpha[im][e];
    __syncthreads();  // Pt + rsum visible
#pragma unroll
    for (int im = 0; im < IM; ++im)
#pragma unroll
      for (int e = 0; e < 8; ++e) {
        float tot = psum[im][e] + rsum[wc ^ 1][wr * 32 + im * 16 + hf * 8 + e];
        lrow[im][e] = lrow[im][e] * alpha[im][e] + tot;
      }
    // ---- O += P @ V (4 k-steps over j)
#pragma unroll
    for (int kp = 0; kp < 4; ++kp) {
      v16bf pf[IM], vf[INO];
#pragma unroll
      for (int im = 0; im < IM; ++im) {
        const bf16_t* pp = &Pt[wr * 32 + im * 16 + l16][kp * 32 + hf * 8];
#pragma unroll
        for (int e = 0; e < 8; ++e) { pf[im][e] = pp[e]; pf[im][e + 8] = pp[16 + e]; }
      }
#pragma unroll
      for (int io = 0; io < INO; ++io) {
        const bf16_t* vp = &Vt[cur][wc * 32 + io * 16 + l16][kp * 32 + hf * 8];
#pragma unroll
        for (int e = 0; e < 8; ++e) { vf[io][e] = vp[e]; vf[io][e + 8] = vp[16 + e]; }
      }
#pragma unroll
      for (int im = 0; im < IM; ++im)
#pragma unroll
        for (int io = 0; io < INO; ++io)
          acco[im][io] = __builtin_amdgcn_wmma_f32_16x16x32_bf16(
              false, pf[im], false, vf[io], (short)0, acco[im][io], false, false);
    }
    if (jt + 1 < KLEN / BJ) wait_async_0();
    __syncthreads();  // Pt reusable; K/V buffers swapped
  }
  // ---- epilogue: O / l -> av bf16 in (i, b, n*64+d)
#pragma unroll
  for (int im = 0; im < IM; ++im) {
#pragma unroll
    for (int io = 0; io < INO; ++io) {
      const int d = wc * 32 + io * 16 + l16;
#pragma unroll
      for (int e = 0; e < 8; ++e) {
        const int i = i0 + wr * 32 + im * 16 + hf * 8 + e;
        av[((long)i * BSZ + b) * NHD + n * 64 + d] =
            f2bf(acco[im][io][e] / lrow[im][e]);
      }
    }
  }
}

// ---------------------------------------------------------------------------
extern "C" void kernel_launch(void* const* d_in, const int* in_sizes, int n_in,
                              void* d_out, int out_size, void* d_ws, size_t ws_size,
                              hipStream_t stream) {
  (void)in_sizes; (void)n_in; (void)out_size; (void)ws_size;
  const float* w    = (const float*)d_in[0];
  const float* r    = (const float*)d_in[1];
  const float* mems = (const float*)d_in[2];
  const float* rwb  = (const float*)d_in[3];
  const float* rrb  = (const float*)d_in[4];
  const int*  amask = (const int*)d_in[5];
  const float* qkvk = (const float*)d_in[6];
  const float* rker = (const float*)d_in[7];
  const float* oker = (const float*)d_in[8];
  float* out = (float*)d_out;

  char* p = (char*)d_ws;
  auto alloc = [&](size_t bytes) -> char* {
    char* q = p;
    p += (bytes + 255) & ~(size_t)255;
    return q;
  };
  bf16_t* cat_bf  = (bf16_t*)alloc((size_t)KLEN * BSZ * DM * 2);
  bf16_t* qkvk_t  = (bf16_t*)alloc((size_t)QKV3 * DM * 2);
  bf16_t* r_bf    = (bf16_t*)alloc((size_t)RLEN * DM * 2);
  bf16_t* rker_t  = (bf16_t*)alloc((size_t)NHD * DM * 2);
  bf16_t* oker_t  = (bf16_t*)alloc((size_t)DM * NHD * 2);
  bf16_t* wheads  = (bf16_t*)alloc((size_t)KLEN * BSZ * QKV3 * 2);  // bf16 now
  bf16_t* rhead   = (bf16_t*)alloc((size_t)RLEN * NHD * 2);         // bf16 now
  bf16_t* qw_bf   = (bf16_t*)alloc((size_t)BSZ * NH * QLEN * DH * 2);
  bf16_t* qr_bf   = (bf16_t*)alloc((size_t)BSZ * NH * QLEN * DH * 2);
  bf16_t* k_bf    = (bf16_t*)alloc((size_t)BSZ * NH * KLEN * DH * 2);
  bf16_t* v_t     = (bf16_t*)alloc((size_t)BSZ * NH * DH * KLEN * 2);
  bf16_t* rk_bf   = (bf16_t*)alloc((size_t)NH * RLEN * DH * 2);
  bf16_t* bdr     = (bf16_t*)alloc((size_t)BSZ * NH * QLEN * RLEN * 2);  // 134MB
  bf16_t* av_bf   = (bf16_t*)alloc((size_t)QLEN * BSZ * NHD * 2);

  // --- bf16 conversions (weights transposed so all GEMMs are trans-B)
  cat_to_bf16_kernel<<<(KLEN * BSZ * DM) / 256, 256, 0, stream>>>(mems, w, cat_bf);
  f32_to_bf16_kernel<<<(RLEN * DM) / 256, 256, 0, stream>>>(r, r_bf, (long)RLEN * DM);
  transpose_f32_to_bf16<<<dim3(QKV3 / 32, DM / 32, 1), 256, 0, stream>>>(qkvk, qkvk_t, DM, QKV3);
  transpose_f32_to_bf16<<<dim3(NHD / 32, DM / 32, 1), 256, 0, stream>>>(rker, rker_t, DM, NHD);
  transpose_f32_to_bf16<<<dim3(DM / 32, NHD / 32, 1), 256, 0, stream>>>(oker, oker_t, NHD, DM);

  // --- G1: w_heads = cat @ qkv_kernel  [8192x3072x1024], bf16 out
  gemm_bf16_wmma<128, 128, 4, 2, 3><<<dim3(QKV3 / 128, (KLEN * BSZ) / 128, 1), 256, 0, stream>>>(
      cat_bf, qkvk_t, (float*)wheads, nullptr, DM, DM, DM, QKV3, 0, 0, 1, 1, 0, 0);
  // --- G2: r_head_k = r @ r_kernel     [1024x1024x1024], bf16 out
  gemm_bf16_wmma<128, 128, 4, 2, 3><<<dim3(NHD / 128, RLEN / 128, 1), 256, 0, stream>>>(
      r_bf, rker_t, (float*)rhead, nullptr, DM, DM, DM, NHD, 0, 0, 1, 1, 0, 0);

  // --- split/bias/relayout
  pack_qkv_kernel<<<(KLEN * BSZ * NHD) / 256, 256, 0, stream>>>(wheads, rwb, rrb,
                                                               qw_bf, qr_bf, k_bf, v_t);
  pack_rk_kernel<<<(RLEN * NHD) / 256, 256, 0, stream>>>(rhead, rk_bf);

  // --- G4: BD_raw[z] = qr[z] @ rk[n]^T, stored bf16 (gathered by fused attn)
  gemm_bf16_wmma<128, 128, 4, 2, 3><<<dim3(RLEN / 128, QLEN / 128, BSZ * NH), 256, 0, stream>>>(
      qr_bf, rk_bf, (float*)bdr, nullptr, DH, DH, DH, RLEN,
      (long)QLEN * DH, (long)RLEN * DH, NH, 1, (long)QLEN * RLEN, 0);

  // --- fused attention: AC + rel_shift(BD) + softmax + P@V -> av bf16
  attn_fused_kernel<<<dim3(QLEN / 128, BSZ * NH, 1), 256, 0, stream>>>(
      qw_bf, k_bf, v_t, bdr, amask, av_bf);

  // --- G6: out = attn_vec @ out_kernel + w   [4096x1024x1024]
  gemm_bf16_wmma<128, 128, 4, 2, 1><<<dim3(DM / 128, (QLEN * BSZ) / 128, 1), 256, 0, stream>>>(
      av_bf, oker_t, out, w, NHD, NHD, DM, DM, 0, 0, 1, 1, 0, 0);
}
